// Attention_22273700397249
// MI455X (gfx1250) — matrix operand order repaired and checked by reading, not compiled, not run on hardware
//
#include <hip/hip_runtime.h>

#define SEQ 8192
#define DIM 512

typedef __attribute__((ext_vector_type(16))) _Float16 v16h;
typedef __attribute__((ext_vector_type(8)))  float    v8f;

__device__ __forceinline__ v8f wmma_f16f32(v16h a, v16h b, v8f c) {
  // D = A(16x32 f16) * B(32x16 f16) + C(16x16 f32)
  return __builtin_amdgcn_wmma_f32_16x16x32_f16(false, a, false, b, (short)0, c,
                                                false, false);
}

// ---- A-matrix loader: 16x32 (MxK) tile.  Per ISA 7.12.2 (16-bit A):
// lanes 0-15: M=lane, K = 0..7 (v0-3) and 16..23 (v4-7)
// lanes 16-31: M=lane-16, K = 8..15 and 24..31
__device__ __forceinline__ v16h load_A_f32(const float* __restrict__ src, int ld,
                                           int row0, int k0, int lane) {
  const int m  = lane & 15;
  const int kb = (lane >> 4) << 3;            // 0 or 8
  const float* p = src + (size_t)(row0 + m) * ld + k0 + kb;
  v16h r;
#pragma unroll
  for (int i = 0; i < 8; ++i) r[i] = (_Float16)p[i];
#pragma unroll
  for (int i = 0; i < 8; ++i) r[8 + i] = (_Float16)p[16 + i];
  return r;
}

__device__ __forceinline__ v16h load_A_f16(const _Float16* __restrict__ src, int ld,
                                           int row0, int k0, int lane) {
  const int m  = lane & 15;
  const int kb = (lane >> 4) << 3;
  const _Float16* p = src + (size_t)(row0 + m) * ld + k0 + kb;
  v16h r;
#pragma unroll
  for (int i = 0; i < 8; ++i) r[i] = p[i];
#pragma unroll
  for (int i = 0; i < 8; ++i) r[8 + i] = p[16 + i];
  return r;
}

// ---- B-matrix loader: 32x16 (KxN) tile, same K order as A:
// lanes 0-15: N=lane, K = 0..7 and 16..23; lanes 16-31: N=lane-16, K = 8..15 and 24..31.
// Source stored N-major: src[n][k].
__device__ __forceinline__ v16h load_B_nk_f32(const float* __restrict__ src, int ld,
                                              int n0, int k0, int lane) {
  const int n  = lane & 15;
  const int kb = (lane >> 4) << 3;            // 0 or 8
  const float* p = src + (size_t)(n0 + n) * ld + k0 + kb;
  v16h r;
#pragma unroll
  for (int i = 0; i < 8; ++i) r[i] = (_Float16)p[i];
#pragma unroll
  for (int i = 0; i < 8; ++i) r[8 + i] = (_Float16)p[16 + i];
  return r;
}

__device__ __forceinline__ v16h load_B_nk_f16(const _Float16* __restrict__ src, int ld,
                                              int n0, int k0, int lane) {
  const int n  = lane & 15;
  const int kb = (lane >> 4) << 3;
  const _Float16* p = src + (size_t)(n0 + n) * ld + k0 + kb;
  v16h r;
#pragma unroll
  for (int i = 0; i < 8; ++i) r[i] = p[i];
#pragma unroll
  for (int i = 0; i < 8; ++i) r[8 + i] = p[16 + i];
  return r;
}

// ---- C/D store: VGPR j, lanes 0-15 -> M=j, lanes 16-31 -> M=j+8; N = lane&15.
__device__ __forceinline__ void store_C_f32(float* __restrict__ dst, int ld,
                                            int row0, int col0, int lane, v8f c) {
  const int n  = lane & 15;
  const int mb = (lane >> 4) << 3;
#pragma unroll
  for (int j = 0; j < 8; ++j)
    dst[(size_t)(row0 + j + mb) * ld + col0 + n] = c[j];
}

// ========== Kernel 1: QKV projection (x @ W^T + b) -> f16, 16x64 per wave ======
__global__ __launch_bounds__(32)
void qkv_kernel(const float* __restrict__ x,
                const float* __restrict__ Wq, const float* __restrict__ bq,
                const float* __restrict__ Wk, const float* __restrict__ bk,
                const float* __restrict__ Wv, const float* __restrict__ bv,
                _Float16* __restrict__ Qh, _Float16* __restrict__ Kh,
                _Float16* __restrict__ Vh) {
  const int lane = threadIdx.x & 31;
  const int row0 = blockIdx.x * 16;           // 512 row tiles
  const int tj   = blockIdx.y;                // 0..23
  const int which = tj >> 3;                  // 0=Q 1=K 2=V
  const int col0  = (tj & 7) * 64;            // 64-wide output slab

  const float* W = (which == 0) ? Wq : (which == 1) ? Wk : Wv;
  const float* b = (which == 0) ? bq : (which == 1) ? bk : bv;
  _Float16*    O = (which == 0) ? Qh : (which == 1) ? Kh : Vh;

  v8f acc[4];
#pragma unroll
  for (int jt = 0; jt < 4; ++jt) acc[jt] = (v8f){};

  for (int k0 = 0; k0 < DIM; k0 += 32) {
    v16h a = load_A_f32(x, DIM, row0, k0, lane);
#pragma unroll
    for (int jt = 0; jt < 4; ++jt) {
      v16h bm = load_B_nk_f32(W, DIM, col0 + jt * 16, k0, lane); // B[k][n]=W[n][k]
      acc[jt] = wmma_f16f32(a, bm, acc[jt]);
    }
  }
  const int n  = lane & 15;
  const int mb = (lane >> 4) << 3;
#pragma unroll
  for (int jt = 0; jt < 4; ++jt) {
    const float bias = b[col0 + jt * 16 + n];
#pragma unroll
    for (int j = 0; j < 8; ++j)
      O[(size_t)(row0 + j + mb) * DIM + col0 + jt * 16 + n] =
          (_Float16)(acc[jt][j] + bias);
  }
}

// ========== Kernel 1b: transpose V -> Vt[512][8192] f16 (LDS tile transpose) ===
__global__ __launch_bounds__(256)
void vtrans_kernel(const _Float16* __restrict__ Vh, _Float16* __restrict__ Vt) {
  __shared__ _Float16 T[64][65];              // +1 pad to dodge bank conflicts
  const int s0 = blockIdx.x * 64;             // sequence tile
  const int d0 = blockIdx.y * 64;             // feature tile
  const int r   = threadIdx.x >> 2;           // 0..63
  const int seg = (threadIdx.x & 3) * 16;     // 0,16,32,48
#pragma unroll
  for (int i = 0; i < 16; ++i)
    T[r][seg + i] = Vh[(size_t)(s0 + r) * DIM + d0 + seg + i];
  __syncthreads();
#pragma unroll
  for (int i = 0; i < 16; ++i)
    Vt[(size_t)(d0 + r) * SEQ + s0 + seg + i] = T[seg + i][r];
}

// ========== Kernel 2: raw scores = Q K^T / sqrt(D), 32x64 per wave =============
__global__ __launch_bounds__(32)
void scores_kernel(const _Float16* __restrict__ Qh, const _Float16* __restrict__ Kh,
                   float* __restrict__ Sc) {
  const int row0 = blockIdx.x * 32;           // 256 row tiles
  const int col0 = blockIdx.y * 64;           // 128 col tiles
  if (col0 > row0 + 31) return;               // tile fully above diagonal
  const int lane = threadIdx.x & 31;

  v8f acc[2][4];
#pragma unroll
  for (int ai = 0; ai < 2; ++ai)
#pragma unroll
    for (int jt = 0; jt < 4; ++jt) acc[ai][jt] = (v8f){};

  for (int k0 = 0; k0 < DIM; k0 += 32) {
    v16h a0 = load_A_f16(Qh, DIM, row0, k0, lane);
    v16h a1 = load_A_f16(Qh, DIM, row0 + 16, k0, lane);
#pragma unroll
    for (int jt = 0; jt < 4; ++jt) {
      v16h b = load_B_nk_f16(Kh, DIM, col0 + jt * 16, k0, lane); // B[k][n]=K[n][k]
      acc[0][jt] = wmma_f16f32(a0, b, acc[0][jt]);
      acc[1][jt] = wmma_f16f32(a1, b, acc[1][jt]);
    }
  }
  const float scale = 0.04419417382415922f;   // 1/sqrt(512)
  const int n  = lane & 15;
  const int mb = (lane >> 4) << 3;
#pragma unroll
  for (int ai = 0; ai < 2; ++ai)
#pragma unroll
    for (int jt = 0; jt < 4; ++jt)
#pragma unroll
      for (int j = 0; j < 8; ++j) {
        const int row = row0 + ai * 16 + j + mb;
        const int col = col0 + jt * 16 + n;
        float v = acc[ai][jt][j] * scale;
        if (col > row) v = -__builtin_inff();
        Sc[(size_t)row * SEQ + col] = v;
      }
}

// ========== Kernel 3: per-row online (max, sum_exp) ============================
__global__ __launch_bounds__(256)
void rowstats_kernel(const float* __restrict__ Sc,
                     float* __restrict__ rowmax, float* __restrict__ rowsum) {
  const int r   = blockIdx.x;
  const int tid = threadIdx.x;
  const float* row = Sc + (size_t)r * SEQ;
  const int nvalid = r + 1;

  float m = -__builtin_inff(), s = 0.f;
  for (int c = tid; c < nvalid; c += 256) {
    const float v = row[c];
    if (v > m) { s *= __expf(m - v); m = v; }
    s += __expf(v - m);
  }
  __shared__ float sm[256], ss[256];
  sm[tid] = m; ss[tid] = s;
  __syncthreads();
  for (int off = 128; off > 0; off >>= 1) {
    if (tid < off) {
      const float m1 = sm[tid], s1 = ss[tid];
      const float m2 = sm[tid + off], s2 = ss[tid + off];
      const float M = fmaxf(m1, m2);
      float Snew = 0.f;
      if (M != -__builtin_inff())
        Snew = s1 * __expf(m1 - M) + s2 * __expf(m2 - M);
      sm[tid] = M; ss[tid] = Snew;
    }
    __syncthreads();
  }
  if (tid == 0) { rowmax[r] = sm[0]; rowsum[r] = ss[0]; }
}

// ========== Kernel 4: normalize + zero upper triangle ==========================
__global__ __launch_bounds__(256)
void norm_kernel(float* __restrict__ Sc,
                 const float* __restrict__ rowmax, const float* __restrict__ rowsum) {
  const int r = blockIdx.x;
  const int c = blockIdx.y * 256 + threadIdx.x;
  const size_t idx = (size_t)r * SEQ + c;
  float w = 0.f;
  if (c <= r) {
    const float m   = rowmax[r];
    const float inv = 1.0f / rowsum[r];       // >= 1 always (diagonal exp(0) term)
    w = __expf(Sc[idx] - m) * inv;
  }
  Sc[idx] = w;
}

// ========== Kernel 5: output = attn @ V; attn strip staged once via LDS ========
// One block per 16-row strip; 8 waves cover all 512 output columns (64 each).
__global__ __launch_bounds__(256)
void av_kernel(const float* __restrict__ Sc, const _Float16* __restrict__ Vt,
               float* __restrict__ Out) {
  __shared__ _Float16 P[16][256];             // f16 attn strip chunk (8 KB)
  const int I    = blockIdx.x;
  const int tid  = threadIdx.x;
  const int lane = tid & 31;
  const int wave = tid >> 5;
  const int row0 = I * 16;
  const int col0 = wave * 64;

  v8f acc[4];
#pragma unroll
  for (int jt = 0; jt < 4; ++jt) acc[jt] = (v8f){};

  const int kend = (((I + 1) * 16) + 255) & ~255;   // zeros past diagonal are safe
  const int lr   = tid >> 4;                  // 0..15 strip row for staging
  const int lseg = (tid & 15) * 16;           // 16-float segment

  for (int kc = 0; kc < kend; kc += 256) {
    __syncthreads();
    {
      const float* src = Sc + (size_t)(row0 + lr) * SEQ + kc + lseg;
#pragma unroll
      for (int i = 0; i < 16; ++i) P[lr][lseg + i] = (_Float16)src[i];
    }
    __syncthreads();
#pragma unroll
    for (int k0 = 0; k0 < 256; k0 += 32) {
      const int m  = lane & 15;
      const int kb = (lane >> 4) << 3;
      v16h a;
#pragma unroll
      for (int i = 0; i < 8; ++i) a[i] = P[m][k0 + kb + i];
#pragma unroll
      for (int i = 0; i < 8; ++i) a[8 + i] = P[m][k0 + kb + 16 + i];
#pragma unroll
      for (int jt = 0; jt < 4; ++jt) {
        // B[k][n] = V[k][n] = Vt[n][k]  (two contiguous 16B runs per lane)
        v16h b = load_B_nk_f16(Vt, SEQ, col0 + jt * 16, kc + k0, lane);
        acc[jt] = wmma_f16f32(a, b, acc[jt]);
      }
    }
  }
#pragma unroll
  for (int jt = 0; jt < 4; ++jt)
    store_C_f32(Out, DIM, row0, col0 + jt * 16, lane, acc[jt]);
}

// =============================== launch ========================================
extern "C" void kernel_launch(void* const* d_in, const int* in_sizes, int n_in,
                              void* d_out, int out_size, void* d_ws, size_t ws_size,
                              hipStream_t stream) {
  const float* x  = (const float*)d_in[0];
  const float* Wq = (const float*)d_in[1];
  const float* bq = (const float*)d_in[2];
  const float* Wk = (const float*)d_in[3];
  const float* bk = (const float*)d_in[4];
  const float* Wv = (const float*)d_in[5];
  const float* bv = (const float*)d_in[6];

  float* out  = (float*)d_out;                      // [8192, 512]
  float* attn = out + (size_t)SEQ * DIM;            // [8192, 8192]

  _Float16* Qh = (_Float16*)d_ws;                   // 8 MB each
  _Float16* Kh = Qh + (size_t)SEQ * DIM;
  _Float16* Vh = Kh + (size_t)SEQ * DIM;
  _Float16* Vt = Vh + (size_t)SEQ * DIM;            // transposed V [512][8192]
  float* rowmax = (float*)(Vt + (size_t)SEQ * DIM);
  float* rowsum = rowmax + SEQ;

  qkv_kernel<<<dim3(SEQ / 16, 24), 32, 0, stream>>>(x, Wq, bq, Wk, bk, Wv, bv,
                                                    Qh, Kh, Vh);
  vtrans_kernel<<<dim3(SEQ / 64, DIM / 64), 256, 0, stream>>>(Vh, Vt);
  scores_kernel<<<dim3(SEQ / 32, SEQ / 64), 32, 0, stream>>>(Qh, Kh, attn);
  rowstats_kernel<<<SEQ, 256, 0, stream>>>(attn, rowmax, rowsum);
  norm_kernel<<<dim3(SEQ, SEQ / 256), 256, 0, stream>>>(attn, rowmax, rowsum);
  av_kernel<<<SEQ / 16, 256, 0, stream>>>(attn, Vt, out);
}
